// Model_74448963109565
// MI455X (gfx1250) — matrix operand order repaired and checked by reading, not compiled, not run on hardware
//
#include <hip/hip_runtime.h>
#include <hip/hip_bf16.h>

// ---------------------------------------------------------------------------
// 6-layer LSTM (H=256, B=64, T=2048, D=128) + FC for MI455X (gfx1250).
//
//  * bf16 WMMA (v_wmma_f32_16x16x32_bf16) with f32 accumulation.
//  * Layer-0 input GEMM (x @ W_ih0^T + b0) precomputed by a WMMA GEMM kernel.
//  * One persistent pipelined kernel: 24 blocks = 6 layers x 4 column-quarters.
//    Each block keeps its 256x512 bf16 slice of [W_ih | W_hh] in LDS (256 KB)
//    plus the per-step activation tile [64 x 512] bf16 (64 KB) -> 320 KB LDS,
//    i.e. one full CDNA5 WGP. Cell state c stays in VGPRs across all T steps.
//  * Layers run in wavefront order, synchronized with per-layer step counters
//    (agent-scope acquire/release atomics, s_sleep spin).
//  * LDS fills use GLOBAL_LOAD_ASYNC_TO_LDS_B128 (ASYNCcnt) — builtin takes
//    (v4i AS1*, v4i AS3*, imm offset, imm cpol) per the round-2 diagnostic.
// ---------------------------------------------------------------------------

typedef unsigned short u16;
typedef __attribute__((ext_vector_type(16))) __bf16 v16bf;
typedef __attribute__((ext_vector_type(8)))  float  v8f;
typedef __attribute__((__vector_size__(16))) int    v4i_;

union Frag16 { v16bf bf; uint4 u[2]; };

#define T_STEPS 2048
#define BATCH   64
#define DIN     128
#define HID     256
#define G4      1024   // 4*HID

#if defined(__gfx1250__) && __has_builtin(__builtin_amdgcn_global_load_async_to_lds_b128)
#define ASYNC_LDS 1
#endif

__device__ __forceinline__ void copy16(const u16* src, u16* dst_lds) {
#ifdef ASYNC_LDS
    __builtin_amdgcn_global_load_async_to_lds_b128(
        (__attribute__((address_space(1))) v4i_*)src,
        (__attribute__((address_space(3))) v4i_*)dst_lds, 0, 0);
#else
    *(uint4*)dst_lds = *(const uint4*)src;
#endif
}

__device__ __forceinline__ void async_wait0() {
#ifdef ASYNC_LDS
#if __has_builtin(__builtin_amdgcn_s_wait_asynccnt)
    __builtin_amdgcn_s_wait_asynccnt(0);
#else
    asm volatile("s_wait_asynccnt 0" ::: "memory");
#endif
#endif
}

__device__ __forceinline__ u16 f2bf(float f) {
    unsigned u = __float_as_uint(f);
    unsigned r = (u + 0x7FFFu + ((u >> 16) & 1u)) >> 16;
    return (u16)r;
}
__device__ __forceinline__ float bf2f(u16 b) {
    return __uint_as_float(((unsigned)b) << 16);
}
__device__ __forceinline__ float fsig(float x) { return 1.0f / (1.0f + __expf(-x)); }
__device__ __forceinline__ float ftanh(float x) { return 2.0f * fsig(2.0f * x) - 1.0f; }

// ---------------------------------------------------------------- utilities
__global__ void k_zero_u32(unsigned* p, int n) {
    int i = blockIdx.x * blockDim.x + threadIdx.x;
    if (i < n) p[i] = 0u;
}

__global__ void k_cvt_bf16(const float* __restrict__ s, u16* __restrict__ d, int n) {
    int i = blockIdx.x * 256 + threadIdx.x;
    if (i < n) d[i] = f2bf(s[i]);
}

__global__ void k_bias_add(const float* __restrict__ a, const float* __restrict__ b,
                           float* __restrict__ d, int n) {
    int i = blockIdx.x * 256 + threadIdx.x;
    if (i < n) d[i] = a[i] + b[i];
}

// x: [B, T, D] f32  ->  xb: [T*B, D] bf16  (time-major rows for the scan)
__global__ void k_cvt_x(const float* __restrict__ x, u16* __restrict__ xb) {
    int i = blockIdx.x * 256 + threadIdx.x;           // i < T*B*D
    int d = i % DIN;
    int r = i / DIN;                                   // r = t*B + b
    int b = r % BATCH;
    int t = r / BATCH;
    xb[i] = f2bf(x[((size_t)b * T_STEPS + t) * DIN + d]);
}

// ------------------------------------------------- layer-0 input GEMM (WMMA)
// out[T*B, 1024] = xb[T*B, 128] @ W^T + bias   (W: [1024,128] bf16 row-major)
__global__ __launch_bounds__(256) void k_gemm_wx0(const u16* __restrict__ xb,
                                                  const u16* __restrict__ W,
                                                  const float* __restrict__ bias,
                                                  float* __restrict__ out) {
    __shared__ u16 sA[32 * 128];
    const int tid  = threadIdx.x;
    const int row0 = blockIdx.x * 32;

    for (int idx = tid * 8; idx < 32 * 128; idx += 256 * 8)
        *(uint4*)&sA[idx] = *(const uint4*)&xb[(size_t)row0 * 128 + idx];
    __syncthreads();

    const int w = tid >> 5, lane = tid & 31, hi = lane >> 4, ln = lane & 15;
    const int cb = blockIdx.y * 512 + w * 64;

    v8f acc[2][4];
    #pragma unroll
    for (int nt = 0; nt < 4; ++nt) {
        float bv = bias[cb + nt * 16 + ln];
        #pragma unroll
        for (int mt = 0; mt < 2; ++mt)
            #pragma unroll
            for (int v = 0; v < 8; ++v) acc[mt][nt][v] = bv;
    }

    #pragma unroll
    for (int kk = 0; kk < 4; ++kk) {
        Frag16 a[2], bfr[4];
        #pragma unroll
        for (int mt = 0; mt < 2; ++mt) {
            const u16* p = &sA[(mt * 16 + ln) * 128 + kk * 32 + hi * 8];
            a[mt].u[0] = *(const uint4*)p;
            a[mt].u[1] = *(const uint4*)(p + 16);
        }
        #pragma unroll
        for (int nt = 0; nt < 4; ++nt) {
            const u16* p = &W[(size_t)(cb + nt * 16 + ln) * 128 + kk * 32 + hi * 8];
            bfr[nt].u[0] = *(const uint4*)p;
            bfr[nt].u[1] = *(const uint4*)(p + 16);
        }
        #pragma unroll
        for (int mt = 0; mt < 2; ++mt)
            #pragma unroll
            for (int nt = 0; nt < 4; ++nt)
                acc[mt][nt] = __builtin_amdgcn_wmma_f32_16x16x32_bf16(
                    false, a[mt].bf, false, bfr[nt].bf, (short)0, acc[mt][nt],
                    false, false);
    }

    #pragma unroll
    for (int mt = 0; mt < 2; ++mt)
        #pragma unroll
        for (int nt = 0; nt < 4; ++nt) {
            float* p = &out[(size_t)(row0 + mt * 16 + 8 * hi) * G4 + cb + nt * 16 + ln];
            #pragma unroll
            for (int v = 0; v < 8; ++v) p[(size_t)v * G4] = acc[mt][nt][v];
        }
}

// ------------------------------------------------ pipelined persistent LSTM
// K and the layer-0 flag are template parameters so the k-loop fully unrolls
// and all index div/mod fold to shifts in the 2048-iteration hot loop.
template <int K, bool L0>
__device__ void lstm_body(int l, int qb, u16* lW, u16* lA,
                          u16* __restrict__ hs_l, const u16* __restrict__ hs_p,
                          const u16* __restrict__ h0b, const float* __restrict__ c0,
                          const u16* __restrict__ Wih, const u16* __restrict__ Whh,
                          const float* __restrict__ biasb,
                          const float* __restrict__ wx0,
                          unsigned* __restrict__ cnt) {
    const int tid = threadIdx.x;

    // one-time weight slice preload: local row r = qg*64 + j
    for (int idx = tid * 8; idx < 256 * K; idx += 256 * 8) {
        int r = idx / K, k = idx % K;                       // shifts (K: pow2)
        int gr = (r >> 6) * 256 + qb * 64 + (r & 63);
        const u16* src;
        if (L0)            src = &Whh[(size_t)gr * HID + k];
        else if (k < 256)  src = &Wih[(size_t)gr * HID + k];
        else               src = &Whh[(size_t)gr * HID + (k - 256)];
        copy16(src, &lW[idx]);
    }
    async_wait0();

    const int w = tid >> 5, lane = tid & 31, hi = lane >> 4, ln = lane & 15;
    int mt[2], jt[2], colH[2];
    v8f cst[2];                               // persistent cell state tiles
    #pragma unroll
    for (int p = 0; p < 2; ++p) {
        int pair = w + p * 8;                 // 16 (mtile,jtile) pairs / 8 waves
        mt[p] = pair >> 2; jt[p] = pair & 3;
        colH[p] = qb * 64 + jt[p] * 16 + ln;
        const float* pc = &c0[((size_t)l * BATCH + mt[p] * 16 + 8 * hi) * HID + colH[p]];
        #pragma unroll
        for (int v = 0; v < 8; ++v) cst[p][v] = pc[(size_t)v * HID];
    }

    for (int t = 0; t < T_STEPS; ++t) {
        if (tid == 0) {
            if (!L0) {                         // need h_{l-1}(t)
                unsigned tgt = 4u * (unsigned)(t + 1);
                while (__hip_atomic_load(&cnt[l - 1], __ATOMIC_ACQUIRE,
                                         __HIP_MEMORY_SCOPE_AGENT) < tgt)
                    __builtin_amdgcn_s_sleep(2);
            }
            if (t > 0) {                       // need full h_l(t-1)
                unsigned tgt = 4u * (unsigned)t;
                while (__hip_atomic_load(&cnt[l], __ATOMIC_ACQUIRE,
                                         __HIP_MEMORY_SCOPE_AGENT) < tgt)
                    __builtin_amdgcn_s_sleep(2);
            }
        }
        __syncthreads();

        // activation tile: [h_{l-1}(t) | h_l(t-1)] (or h0 at t==0)
        for (int idx = tid * 8; idx < 64 * K; idx += 256 * 8) {
            int b = idx / K, k = idx % K;
            const u16* src;
            if (L0) {
                src = (t == 0) ? &h0b[((size_t)l * BATCH + b) * HID + k]
                               : &hs_l[((size_t)(t - 1) * BATCH + b) * HID + k];
            } else if (k < 256) {
                src = &hs_p[((size_t)t * BATCH + b) * HID + k];
            } else {
                int k2 = k - 256;
                src = (t == 0) ? &h0b[((size_t)l * BATCH + b) * HID + k2]
                               : &hs_l[((size_t)(t - 1) * BATCH + b) * HID + k2];
            }
            copy16(src, &lA[idx]);
        }
        async_wait0();
        __syncthreads();

        constexpr int ksteps = K >> 5;
        #pragma unroll
        for (int p = 0; p < 2; ++p) {
            v8f acc[4];                       // gate tiles i,f,g,o
            if (L0) {
                const float* pw = &wx0[((size_t)t * BATCH + mt[p] * 16 + 8 * hi) * G4];
                #pragma unroll
                for (int qg = 0; qg < 4; ++qg) {
                    int col = qg * 256 + colH[p];
                    #pragma unroll
                    for (int v = 0; v < 8; ++v) acc[qg][v] = pw[(size_t)v * G4 + col];
                }
            } else {
                const float* pb = &biasb[l * G4];
                #pragma unroll
                for (int qg = 0; qg < 4; ++qg) {
                    float bv = pb[qg * 256 + colH[p]];
                    #pragma unroll
                    for (int v = 0; v < 8; ++v) acc[qg][v] = bv;
                }
            }
            #pragma unroll
            for (int kk = 0; kk < ksteps; ++kk) {
                Frag16 a;
                const u16* pa = &lA[(mt[p] * 16 + ln) * K + kk * 32 + hi * 8];
                a.u[0] = *(const uint4*)pa;
                a.u[1] = *(const uint4*)(pa + 16);
                #pragma unroll
                for (int qg = 0; qg < 4; ++qg) {
                    Frag16 bfr;
                    const u16* pbm = &lW[(qg * 64 + jt[p] * 16 + ln) * K + kk * 32 + hi * 8];
                    bfr.u[0] = *(const uint4*)pbm;
                    bfr.u[1] = *(const uint4*)(pbm + 16);
                    acc[qg] = __builtin_amdgcn_wmma_f32_16x16x32_bf16(
                        false, a.bf, false, bfr.bf, (short)0, acc[qg], false, false);
                }
            }
            // LSTM cell element-wise, c in registers, write h (bf16)
            u16* ph = &hs_l[((size_t)t * BATCH + mt[p] * 16 + 8 * hi) * HID + colH[p]];
            #pragma unroll
            for (int v = 0; v < 8; ++v) {
                float iv = fsig(acc[0][v]);
                float fv = fsig(acc[1][v]);
                float gv = ftanh(acc[2][v]);
                float ov = fsig(acc[3][v]);
                float cc = fv * cst[p][v] + iv * gv;
                cst[p][v] = cc;
                ph[(size_t)v * HID] = f2bf(ov * ftanh(cc));
            }
        }

        __threadfence();
        __syncthreads();
        if (tid == 0)
            __hip_atomic_fetch_add(&cnt[l], 1u, __ATOMIC_RELEASE,
                                   __HIP_MEMORY_SCOPE_AGENT);
    }
}

__global__ __launch_bounds__(256, 1) void k_lstm_pipeline(
    u16* __restrict__ hsbase,            // 6 x [T*B, H] bf16
    const u16* __restrict__ h0b,         // [6, 64, 256] bf16
    const float* __restrict__ c0,        // [6, 64, 256] f32
    const u16* __restrict__ wihb,        // [5, 1024, 256] bf16 (layers 1..5)
    const u16* __restrict__ whhb,        // [6, 1024, 256] bf16
    const float* __restrict__ biasb,     // [6, 1024] f32 (b_ih + b_hh)
    const float* __restrict__ wx0,       // [T*B, 1024] f32
    unsigned* __restrict__ cnt) {
    extern __shared__ u16 smem[];
    const int l  = blockIdx.x >> 2;
    const int qb = blockIdx.x & 3;

    u16*       hs_l = hsbase + (size_t)l * T_STEPS * BATCH * HID;
    const u16* hs_p = (l > 0) ? hsbase + (size_t)(l - 1) * T_STEPS * BATCH * HID : nullptr;
    const u16* Whh  = whhb + (size_t)l * G4 * HID;
    const u16* Wih  = (l > 0) ? wihb + (size_t)(l - 1) * G4 * HID : nullptr;

    if (l == 0)
        lstm_body<256, true >(l, qb, smem, smem + 256 * 256, hs_l, hs_p, h0b, c0,
                              Wih, Whh, biasb, wx0, cnt);
    else
        lstm_body<512, false>(l, qb, smem, smem + 256 * 512, hs_l, hs_p, h0b, c0,
                              Wih, Whh, biasb, wx0, cnt);
}

// ---------------------------------------------------------------- final FC
__global__ void k_fc(const u16* __restrict__ h, const float* __restrict__ W,
                     const float* __restrict__ bias, float* __restrict__ out) {
    int b = threadIdx.x;
    int j = blockIdx.x;
    float s = bias[j];
    const u16*   ph = &h[(size_t)b * HID];
    const float* pw = &W[(size_t)j * HID];
    for (int k = 0; k < HID; ++k) s += bf2f(ph[k]) * pw[k];
    out[b * 10 + j] = s;
}

// ---------------------------------------------------------------------------
extern "C" void kernel_launch(void* const* d_in, const int* in_sizes, int n_in,
                              void* d_out, int out_size, void* d_ws, size_t ws_size,
                              hipStream_t stream) {
    (void)in_sizes; (void)n_in; (void)out_size; (void)ws_size;
    const float* x        = (const float*)d_in[0];
    const float* h0       = (const float*)d_in[1];
    const float* c0       = (const float*)d_in[2];
    const float* W_ih0    = (const float*)d_in[3];
    const float* W_hh0    = (const float*)d_in[4];
    const float* b_ih0    = (const float*)d_in[5];
    const float* b_hh0    = (const float*)d_in[6];
    const float* W_ih_r   = (const float*)d_in[7];
    const float* W_hh_r   = (const float*)d_in[8];
    const float* b_ih_r   = (const float*)d_in[9];
    const float* b_hh_r   = (const float*)d_in[10];
    const float* fc_W     = (const float*)d_in[11];
    const float* fc_b     = (const float*)d_in[12];
    float* out = (float*)d_out;

    // workspace layout
    char* ws = (char*)d_ws;
    size_t off = 0;
    auto take = [&](size_t bytes) {
        size_t o = off;
        off += (bytes + 255) & ~(size_t)255;
        return o;
    };
    const size_t TB = (size_t)T_STEPS * BATCH;
    float* wx0   = (float*)(ws + take(TB * G4 * 4));                 // 537 MB
    u16*   xb    = (u16*)  (ws + take(TB * DIN * 2));                // 33.5 MB
    u16*   hsbuf = (u16*)  (ws + take((size_t)6 * TB * HID * 2));    // 402 MB
    u16*   wih0b = (u16*)  (ws + take((size_t)G4 * DIN * 2));
    u16*   wihb  = (u16*)  (ws + take((size_t)5 * G4 * HID * 2));
    u16*   whhb  = (u16*)  (ws + take((size_t)6 * G4 * HID * 2));
    float* biasb = (float*)(ws + take((size_t)6 * G4 * 4));
    u16*   h0b   = (u16*)  (ws + take((size_t)6 * BATCH * HID * 2));
    unsigned* cnt = (unsigned*)(ws + take(64 * 4));

    // reset sync counters every call (deterministic; ws is not re-poisoned)
    k_zero_u32<<<1, 64, 0, stream>>>(cnt, 64);

    // weight / state conversions to bf16
    auto cvt = [&](const float* s, u16* d, int n) {
        k_cvt_bf16<<<(n + 255) / 256, 256, 0, stream>>>(s, d, n);
    };
    cvt(W_ih0, wih0b, G4 * DIN);
    cvt(W_hh0, whhb, G4 * HID);
    cvt(W_ih_r, wihb, 5 * G4 * HID);
    cvt(W_hh_r, whhb + (size_t)G4 * HID, 5 * G4 * HID);
    cvt(h0, h0b, 6 * BATCH * HID);

    k_bias_add<<<(G4 + 255) / 256, 256, 0, stream>>>(b_ih0, b_hh0, biasb, G4);
    k_bias_add<<<(5 * G4 + 255) / 256, 256, 0, stream>>>(b_ih_r, b_hh_r,
                                                         biasb + G4, 5 * G4);

    // x -> bf16 time-major
    {
        int n = (int)(TB * DIN);
        k_cvt_x<<<(n + 255) / 256, 256, 0, stream>>>(x, xb);
    }

    // layer-0 input GEMM: wx0 = xb @ W_ih0^T + (b_ih0 + b_hh0)
    {
        dim3 grid((unsigned)(TB / 32), 2);
        k_gemm_wx0<<<grid, 256, 0, stream>>>(xb, wih0b, biasb, wx0);
    }

    // pipelined 6-layer LSTM scan: 24 persistent blocks, 320 KB LDS each
    k_lstm_pipeline<<<24, 256, 327680, stream>>>(hsbuf, h0b, c0, wihb, whhb,
                                                 biasb, wx0, cnt);

    // final FC on the last timestep of the top layer
    const u16* hlast = hsbuf + (size_t)5 * TB * HID + (size_t)(T_STEPS - 1) * BATCH * HID;
    k_fc<<<10, 64, 0, stream>>>(hlast, fc_W, fc_b, out);
}
